// SBLAAttention_47974784696449
// MI455X (gfx1250) — compile-verified, matrix operand order checked by reading, not run on hardware
//
#include <hip/hip_runtime.h>
#include <cstdint>
#include <cstddef>

// ---------------- problem constants (match reference) ----------------
constexpr int Bc  = 2;       // batch
constexpr int Lc  = 4096;    // seq len
constexpr int Dc  = 2048;    // hidden
constexpr int Hc  = 32;      // heads
constexpr int DHc = 64;      // head dim
constexpr int Mc  = 4;       // landmarks per block
constexpr int Sc  = 64;      // sqrt block
constexpr int NBc = Lc / Sc;     // 64 blocks
constexpr int NLc = NBc * Mc;    // 256 landmarks
constexpr float SCALEc = 0.125f; // DH^-0.5
constexpr float EPSc = 1e-9f;

typedef __attribute__((ext_vector_type(16))) __bf16 bf16x16;
typedef __attribute__((ext_vector_type(8)))  float f32x8;

// ---------------- fp32 -> bf16 (RNE) ----------------
__device__ __forceinline__ unsigned short f2bf(float f) {
  unsigned u = __float_as_uint(f);
  return (unsigned short)((u + 0x7FFFu + ((u >> 16) & 1u)) >> 16);
}

__global__ void k_cast_bf16(const float* __restrict__ src,
                            unsigned short* __restrict__ dst, int n) {
  int i = blockIdx.x * blockDim.x + threadIdx.x;
  if (i < n) dst[i] = f2bf(src[i]);
}

// cast + transpose: src [K,N] fp32 -> dst [N,K] bf16 (K,N multiples of 32)
__global__ __launch_bounds__(256) void k_cast_transpose_bf16(
    const float* __restrict__ src, unsigned short* __restrict__ dst,
    int K, int N) {
  __shared__ float tile[32][33];
  const int bx = blockIdx.x;             // N / 32
  const int by = blockIdx.y;             // K / 32
  const int tx = threadIdx.x & 31, ty = threadIdx.x >> 5;  // 32 x 8
#pragma unroll
  for (int i = 0; i < 4; ++i) {
    const int k = by * 32 + ty + i * 8;
    tile[ty + i * 8][tx] = src[(size_t)k * N + bx * 32 + tx];
  }
  __syncthreads();
#pragma unroll
  for (int i = 0; i < 4; ++i) {
    const int n = bx * 32 + ty + i * 8;
    dst[(size_t)n * K + by * 32 + tx] = f2bf(tile[tx][ty + i * 8]);
  }
}

__global__ void k_add_inplace(float* __restrict__ dst,
                              const float* __restrict__ src, int n) {
  int i = blockIdx.x * blockDim.x + threadIdx.x;
  if (i < n) dst[i] += src[i];
}

// ---------------- bf16 WMMA GEMM: C[M,N] = A[M,K] @ W[K,N] + bias ----------------
// W supplied pre-transposed as WT[N,K]. Block tile 128x128x64, 8 waves;
// wave tile 32x64 (2x4 accumulators of v_wmma_f32_16x16x32_bf16).
// Tiles are filled with global_load_async_to_lds_b128 (ASYNCcnt-tracked).
#define BM 128
#define BN 128
#define BK 64
#define BKP 72  // padded row elems: 144B rows, 16B aligned, bank-staggered

__global__ __launch_bounds__(256) void k_gemm_bf16(
    const unsigned short* __restrict__ A,   // [M,K] bf16 bits
    const unsigned short* __restrict__ WT,  // [N,K] bf16 bits (transposed W)
    const float* __restrict__ bias,         // [N] (may be null)
    float* __restrict__ C,                  // [M,N]
    int Mdim, int N, int K) {
  __shared__ unsigned short As[BM][BKP];
  __shared__ unsigned short Bs[BN][BKP];

  const int tid   = threadIdx.x;
  const int wave  = tid >> 5;
  const int lane  = tid & 31;
  const int lrow  = lane & 15;
  const int lhalf = lane >> 4;

  const int m_blk = blockIdx.y * BM;
  const int n_blk = blockIdx.x * BN;
  const int wm = (wave >> 1) * 32;   // 0,32,64,96
  const int wn = (wave & 1) * 64;    // 0,64

  // per-thread tile-fill coordinates: 2 threads per 64-elem (128B) row
  const int fr = tid >> 1;            // row 0..127
  const int fc = (tid & 1) * 32;      // col 0 or 32
  const unsigned ldsA = (unsigned)(uintptr_t)&As[fr][fc];
  const unsigned ldsB = (unsigned)(uintptr_t)&Bs[fr][fc];
  const unsigned short* gA = A  + (size_t)(m_blk + fr) * K + fc;
  const unsigned short* gB = WT + (size_t)(n_blk + fr) * K + fc;

  f32x8 acc[2][4];
#pragma unroll
  for (int mi = 0; mi < 2; ++mi)
#pragma unroll
    for (int ni = 0; ni < 4; ++ni)
#pragma unroll
      for (int r = 0; r < 8; ++r) acc[mi][ni][r] = 0.0f;

  for (int k0 = 0; k0 < K; k0 += BK) {
    // --- async fill of both tiles: 4x b128 each, straight into LDS ---
#pragma unroll
    for (int i = 0; i < 4; ++i) {
      unsigned long long ga = (unsigned long long)(uintptr_t)(gA + k0 + i * 8);
      asm volatile("global_load_async_to_lds_b128 %0, %1, off"
                   :: "v"(ldsA + i * 16), "v"(ga) : "memory");
    }
#pragma unroll
    for (int i = 0; i < 4; ++i) {
      unsigned long long ga = (unsigned long long)(uintptr_t)(gB + k0 + i * 8);
      asm volatile("global_load_async_to_lds_b128 %0, %1, off"
                   :: "v"(ldsB + i * 16), "v"(ga) : "memory");
    }
    if (k0 + BK < K) {  // warm L2 for the next tiles
      __builtin_prefetch(gA + k0 + BK, 0, 1);
      __builtin_prefetch(gB + k0 + BK, 0, 1);
    }
    asm volatile("s_wait_asynccnt 0" ::: "memory");
    __syncthreads();

    // --- 2 K-steps of 32: build fragments per ISA 16-bit 16x32 layout ---
#pragma unroll
    for (int kk = 0; kk < BK; kk += 32) {
      union Frag { bf16x16 v; unsigned u[8]; };
      Frag a[2], b[4];
#pragma unroll
      for (int mi = 0; mi < 2; ++mi) {
        const unsigned* row = (const unsigned*)&As[wm + mi * 16 + lrow][0];
#pragma unroll
        for (int i = 0; i < 8; ++i) {
          const int kofs = kk + ((i < 4) ? (2 * i + lhalf * 8)
                                         : (16 + 2 * (i - 4) + lhalf * 8));
          a[mi].u[i] = row[kofs >> 1];
        }
      }
#pragma unroll
      for (int ni = 0; ni < 4; ++ni) {
        const unsigned* row = (const unsigned*)&Bs[wn + ni * 16 + lrow][0];
#pragma unroll
        for (int i = 0; i < 8; ++i) {
          const int kofs = kk + ((i < 4) ? (2 * i + lhalf * 8)
                                         : (16 + 2 * (i - 4) + lhalf * 8));
          b[ni].u[i] = row[kofs >> 1];
        }
      }
#pragma unroll
      for (int mi = 0; mi < 2; ++mi)
#pragma unroll
        for (int ni = 0; ni < 4; ++ni)
          acc[mi][ni] = __builtin_amdgcn_wmma_f32_16x16x32_bf16(
              false, a[mi].v, false, b[ni].v, (short)0, acc[mi][ni], false, false);
    }
    __syncthreads();  // protect LDS from next iteration's async writes
  }

  // --- epilogue: C/D layout VGPR r -> M = r + lhalf*8, N = lrow ---
#pragma unroll
  for (int mi = 0; mi < 2; ++mi) {
#pragma unroll
    for (int ni = 0; ni < 4; ++ni) {
      const int n = n_blk + wn + ni * 16 + lrow;
      const float bv = bias ? bias[n] : 0.0f;
#pragma unroll
      for (int r = 0; r < 8; ++r) {
        const int m = m_blk + wm + mi * 16 + r + lhalf * 8;
        C[(size_t)m * N + n] = acc[mi][ni][r] + bv;
      }
    }
  }
}

// ---------------- RoPE over q and k sections of qkv [B*L, 3D] ----------------
__global__ void k_rope(float* __restrict__ qkv) {
  const int idx = blockIdx.x * blockDim.x + threadIdx.x;
  const int total = Bc * Lc * Hc * (DHc / 2) * 2;
  if (idx >= total) return;
  const int sec = idx & 1;     // 0=q, 1=k
  int t = idx >> 1;
  const int i = t & 31; t >>= 5;       // pair index within head
  const int h = t & 31; t >>= 5;       // head
  const int row = t;                   // b*L + l
  const int l = row & (Lc - 1);
  const float inv = __expf(-(2.0f * (float)i / (float)DHc) * 9.210340372f);
  const float fr = (float)l * inv;
  float s, c; __sincosf(fr, &s, &c);
  const size_t base = (size_t)row * (3 * Dc) + (size_t)sec * Dc + h * DHc + 2 * i;
  const float xe = qkv[base], xo = qkv[base + 1];
  qkv[base]     = xe * c - xo * s;
  qkv[base + 1] = xe * s + xo * c;
}

// ---------------- local causal attention within each S-block ----------------
__global__ __launch_bounds__(64) void k_local_attn(
    const float* __restrict__ qkv, float* __restrict__ out_local) {
  const int wg = blockIdx.x;
  const int n = wg % NBc, h = (wg / NBc) % Hc, b = wg / (NBc * Hc);
  __shared__ float kt[Sc][DHc];
  __shared__ float vt[Sc][DHc];
  const int t = threadIdx.x;
  const size_t rowbase = (size_t)(b * Lc + n * Sc + t) * (3 * Dc) + h * DHc;
  for (int d = 0; d < DHc; ++d) {
    kt[t][d] = qkv[rowbase + Dc + d];
    vt[t][d] = qkv[rowbase + 2 * Dc + d];
  }
  __syncthreads();
  float qreg[DHc];
  for (int d = 0; d < DHc; ++d) qreg[d] = qkv[rowbase + d];
  float p[Sc];
  float mx = -INFINITY;
  for (int j = 0; j <= t; ++j) {
    float s = 0.f;
    for (int d = 0; d < DHc; ++d) s += qreg[d] * kt[j][d];
    s *= SCALEc;
    p[j] = s; mx = fmaxf(mx, s);
  }
  float sum = 0.f;
  for (int j = 0; j <= t; ++j) { p[j] = __expf(p[j] - mx); sum += p[j]; }
  const float inv = 1.0f / fmaxf(sum, EPSc);
  const size_t ob = ((size_t)(b * Hc + h) * Lc + n * Sc + t) * DHc;
  for (int d = 0; d < DHc; ++d) {
    float acc = 0.f;
    for (int j = 0; j <= t; ++j) acc += p[j] * vt[j][d];
    out_local[ob + d] = acc * inv;
  }
}

// ---------------- landmark pooling & landmark->token refinement ----------------
__global__ __launch_bounds__(128) void k_pool_l2t(
    const float* __restrict__ qkv, const float* __restrict__ qsrc,
    float* __restrict__ outp, int qmode) {
  const int wg = blockIdx.x;
  const int n = wg % NBc, b = wg / NBc;
  const int t = threadIdx.x, m = t & 3, h = t >> 2;
  float qreg[DHc];
  if (qmode == 0) {
    for (int d = 0; d < DHc; ++d) qreg[d] = qsrc[(size_t)m * Dc + h * DHc + d];
  } else {
    const size_t qb = (size_t)(b * NLc + n * Mc + m) * Dc + h * DHc;
    for (int d = 0; d < DHc; ++d) qreg[d] = qsrc[qb + d];
  }
  float p[Sc]; float mx = -INFINITY;
  for (int j = 0; j < Sc; ++j) {
    const size_t kb = (size_t)(b * Lc + n * Sc + j) * (3 * Dc) + Dc + h * DHc;
    float s = 0.f;
    for (int d = 0; d < DHc; ++d) s += qreg[d] * qkv[kb + d];
    s *= SCALEc; p[j] = s; mx = fmaxf(mx, s);
  }
  float sum = 0.f;
  for (int j = 0; j < Sc; ++j) { p[j] = __expf(p[j] - mx); sum += p[j]; }
  const float inv = 1.0f / sum;
  const size_t ob = (size_t)(b * NLc + n * Mc + m) * Dc + h * DHc;
  for (int d = 0; d < DHc; ++d) {
    float acc = 0.f;
    for (int j = 0; j < Sc; ++j) {
      const size_t vb = (size_t)(b * Lc + n * Sc + j) * (3 * Dc) + 2 * Dc + h * DHc;
      acc += p[j] * qkv[vb + d];
    }
    outp[ob + d] = acc * inv;
  }
}

// ---------------- landmark self-attention (block-causal, online softmax) ----------
__global__ __launch_bounds__(256) void k_lm_attn(
    const float* __restrict__ lqkv, float* __restrict__ lo) {
  const int b = blockIdx.x / Hc, h = blockIdx.x % Hc;
  const int i = threadIdx.x;
  float qreg[DHc];
  const size_t qb = (size_t)(b * NLc + i) * (3 * Dc) + h * DHc;
  for (int d = 0; d < DHc; ++d) qreg[d] = lqkv[qb + d];
  const int lim = ((i >> 2) + 1) * Mc;
  float acc[DHc];
  for (int d = 0; d < DHc; ++d) acc[d] = 0.f;
  float mx = -INFINITY, sum = 0.f;
  for (int j = 0; j < lim; ++j) {
    const size_t kb = (size_t)(b * NLc + j) * (3 * Dc) + Dc + h * DHc;
    float s = 0.f;
    for (int d = 0; d < DHc; ++d) s += qreg[d] * lqkv[kb + d];
    s *= SCALEc;
    const float nm = fmaxf(mx, s);
    const float corr = __expf(mx - nm);
    const float w = __expf(s - nm);
    sum = sum * corr + w;
    const size_t vb = kb + Dc;
    for (int d = 0; d < DHc; ++d) acc[d] = acc[d] * corr + w * lqkv[vb + d];
    mx = nm;
  }
  const float inv = 1.0f / fmaxf(sum, EPSc);
  const size_t ob = (size_t)(b * NLc + i) * Dc + h * DHc;
  for (int d = 0; d < DHc; ++d) lo[ob + d] = acc[d] * inv;
}

// ---------------- token -> landmark global attention ----------------
__global__ __launch_bounds__(256) void k_global_attn(
    const float* __restrict__ qkv, const float* __restrict__ gkv,
    float* __restrict__ out_global) {
  const int idx = blockIdx.x * blockDim.x + threadIdx.x;
  const int l = idx % Lc, h = (idx / Lc) % Hc, b = idx / (Lc * Hc);
  const int lim = (l >> 6) * Mc;  // strictly previous blocks
  const size_t qb = (size_t)(b * Lc + l) * (3 * Dc) + h * DHc;
  float qreg[DHc];
  for (int d = 0; d < DHc; ++d) qreg[d] = qkv[qb + d];
  float acc[DHc];
  for (int d = 0; d < DHc; ++d) acc[d] = 0.f;
  float mx = -INFINITY, sum = 0.f;
  for (int j = 0; j < lim; ++j) {
    const size_t kb = (size_t)(b * NLc + j) * (2 * Dc) + h * DHc;
    float s = 0.f;
    for (int d = 0; d < DHc; ++d) s += qreg[d] * gkv[kb + d];
    s *= SCALEc;
    const float nm = fmaxf(mx, s);
    const float corr = __expf(mx - nm);
    const float w = __expf(s - nm);
    sum = sum * corr + w;
    const size_t vb = kb + Dc;
    for (int d = 0; d < DHc; ++d) acc[d] = acc[d] * corr + w * gkv[vb + d];
    mx = nm;
  }
  const float inv = 1.0f / fmaxf(sum, EPSc);  // lim==0 -> zeros
  const size_t ob = ((size_t)(b * Hc + h) * Lc + l) * DHc;
  for (int d = 0; d < DHc; ++d) out_global[ob + d] = acc[d] * inv;
}

// ---------------- per-head sigmoid gate ----------------
__global__ void k_gate(const float* __restrict__ x, const float* __restrict__ wg,
                       const float* __restrict__ bg, float* __restrict__ gate) {
  const int idx = blockIdx.x * blockDim.x + threadIdx.x;  // row*H + h
  const int h = idx % Hc;
  const int row = idx / Hc;
  float s = bg[h];
  for (int d = 0; d < Dc; ++d) s += x[(size_t)row * Dc + d] * wg[(size_t)d * Hc + h];
  gate[idx] = 1.0f / (1.0f + __expf(-s));
}

// ---------------- gated fusion into [B*L, D] ----------------
__global__ void k_fuse(const float* __restrict__ ol, const float* __restrict__ og,
                       const float* __restrict__ gate, float* __restrict__ fused) {
  const int idx = blockIdx.x * blockDim.x + threadIdx.x;
  const int d = idx % DHc;
  const int h = (idx / DHc) % Hc;
  const int row = idx / Dc;
  const int b = row / Lc, l = row % Lc;
  const float g = gate[(size_t)row * Hc + h];
  const size_t a = ((size_t)(b * Hc + h) * Lc + l) * DHc + d;
  fused[idx] = (1.0f - g) * ol[a] + g * og[a];
}

// ======================= host orchestration =======================
extern "C" void kernel_launch(void* const* d_in, const int* in_sizes, int n_in,
                              void* d_out, int out_size, void* d_ws, size_t ws_size,
                              hipStream_t stream) {
  (void)in_sizes; (void)n_in; (void)out_size; (void)ws_size;
  const float* x        = (const float*)d_in[0];
  const float* w_qkv    = (const float*)d_in[1];
  const float* b_qkv    = (const float*)d_in[2];
  const float* pool_q   = (const float*)d_in[3];
  const float* w_lm_qkv = (const float*)d_in[4];
  const float* b_lm_qkv = (const float*)d_in[5];
  const float* w_lm_out = (const float*)d_in[6];
  const float* b_lm_out = (const float*)d_in[7];
  const float* w_l2t_q  = (const float*)d_in[8];
  const float* b_l2t_q  = (const float*)d_in[9];
  const float* w_l2t_out= (const float*)d_in[10];
  const float* b_l2t_out= (const float*)d_in[11];
  const float* w_lm_kv  = (const float*)d_in[12];
  const float* b_lm_kv  = (const float*)d_in[13];
  const float* w_out    = (const float*)d_in[14];
  const float* b_out    = (const float*)d_in[15];
  const float* w_gate   = (const float*)d_in[16];
  const float* b_gate   = (const float*)d_in[17];
  float* out = (float*)d_out;

  char* ws = (char*)d_ws;
  size_t off = 0;
  auto carve = [&](size_t bytes) -> char* {
    char* p = ws + off;
    off += (bytes + 255) & ~(size_t)255;
    return p;
  };
  const size_t ML = (size_t)Bc * Lc;   // 8192 token rows
  const size_t MR = (size_t)Bc * NLc;  // 512 landmark rows

  unsigned short* XB        = (unsigned short*)carve(ML * Dc * 2);
  unsigned short* WT_QKV    = (unsigned short*)carve((size_t)Dc * 3 * Dc * 2);
  unsigned short* WT_LMQKV  = (unsigned short*)carve((size_t)Dc * 3 * Dc * 2);
  unsigned short* WT_LMOUT  = (unsigned short*)carve((size_t)Dc * Dc * 2);
  unsigned short* WT_L2TQ   = (unsigned short*)carve((size_t)Dc * Dc * 2);
  unsigned short* WT_L2TOUT = (unsigned short*)carve((size_t)Dc * Dc * 2);
  unsigned short* WT_LMKV   = (unsigned short*)carve((size_t)Dc * 2 * Dc * 2);
  unsigned short* WT_OUT    = (unsigned short*)carve((size_t)Dc * Dc * 2);
  float* QKV     = (float*)carve(ML * 3 * Dc * 4);
  float* OUTLOC  = (float*)carve(ML * Dc * 4);
  float* OUTGLB  = (float*)carve(ML * Dc * 4);
  float* LMF     = (float*)carve(MR * Dc * 4);
  unsigned short* LMFB = (unsigned short*)carve(MR * Dc * 2);
  float* LQKV    = (float*)carve(MR * 3 * Dc * 4);
  float* LO      = (float*)carve(MR * Dc * 4);
  unsigned short* LOB = (unsigned short*)carve(MR * Dc * 2);
  float* TMP     = (float*)carve(MR * Dc * 4);
  float* L2Q     = (float*)carve(MR * Dc * 4);
  float* GKV     = (float*)carve(MR * 2 * Dc * 4);
  float* GATE    = (float*)carve(ML * Hc * 4);
  float* FUSED           = QKV;  // alias: qkv dead by fuse time
  unsigned short* FUSEDB = XB;   // alias: xb dead after qkv gemm

  auto cast = [&](const float* s, unsigned short* d, size_t n) {
    k_cast_bf16<<<(unsigned)((n + 255) / 256), 256, 0, stream>>>(s, d, (int)n);
  };
  auto castT = [&](const float* s, unsigned short* d, int K, int N) {
    dim3 grid(N / 32, K / 32);
    k_cast_transpose_bf16<<<grid, 256, 0, stream>>>(s, d, K, N);
  };
  auto gemm = [&](const unsigned short* A, const unsigned short* WT,
                  const float* bias, float* C, int Mdim, int N, int K) {
    dim3 grid(N / BN, Mdim / BM);
    k_gemm_bf16<<<grid, 256, 0, stream>>>(A, WT, bias, C, Mdim, N, K);
  };

  // 1) casts (+ weight transposes so GEMM B-tiles are contiguous 128b loads)
  cast(x, XB, ML * Dc);
  castT(w_qkv,    WT_QKV,    Dc, 3 * Dc);
  castT(w_lm_qkv, WT_LMQKV,  Dc, 3 * Dc);
  castT(w_lm_out, WT_LMOUT,  Dc, Dc);
  castT(w_l2t_q,  WT_L2TQ,   Dc, Dc);
  castT(w_l2t_out,WT_L2TOUT, Dc, Dc);
  castT(w_lm_kv,  WT_LMKV,   Dc, 2 * Dc);
  castT(w_out,    WT_OUT,    Dc, Dc);

  // 2) QKV projection (WMMA)
  gemm(XB, WT_QKV, b_qkv, QKV, (int)ML, 3 * Dc, Dc);

  // 3) RoPE on q,k
  {
    const int total = Bc * Lc * Hc * (DHc / 2) * 2;
    k_rope<<<(total + 255) / 256, 256, 0, stream>>>(QKV);
  }

  // 4) local blocked causal attention
  k_local_attn<<<Bc * Hc * NBc, 64, 0, stream>>>(QKV, OUTLOC);

  // 5) landmark pooling -> LMF
  k_pool_l2t<<<Bc * NBc, 128, 0, stream>>>(QKV, pool_q, LMF, 0);

  // 6) landmark self-attention branch
  cast(LMF, LMFB, MR * Dc);
  gemm(LMFB, WT_LMQKV, b_lm_qkv, LQKV, (int)MR, 3 * Dc, Dc);
  k_lm_attn<<<Bc * Hc, 256, 0, stream>>>(LQKV, LO);
  cast(LO, LOB, MR * Dc);
  gemm(LOB, WT_LMOUT, b_lm_out, TMP, (int)MR, Dc, Dc);
  k_add_inplace<<<(unsigned)((MR * Dc + 255) / 256), 256, 0, stream>>>(LMF, TMP, (int)(MR * Dc));

  // 7) landmark -> token refinement
  cast(LMF, LMFB, MR * Dc);
  gemm(LMFB, WT_L2TQ, b_l2t_q, L2Q, (int)MR, Dc, Dc);
  k_pool_l2t<<<Bc * NBc, 128, 0, stream>>>(QKV, L2Q, LO, 1);
  cast(LO, LOB, MR * Dc);
  gemm(LOB, WT_L2TOUT, b_l2t_out, TMP, (int)MR, Dc, Dc);
  k_add_inplace<<<(unsigned)((MR * Dc + 255) / 256), 256, 0, stream>>>(LMF, TMP, (int)(MR * Dc));

  // 8) landmark K/V
  cast(LMF, LMFB, MR * Dc);
  gemm(LMFB, WT_LMKV, b_lm_kv, GKV, (int)MR, 2 * Dc, Dc);

  // 9) token -> landmark global attention
  k_global_attn<<<(Bc * Hc * Lc) / 256, 256, 0, stream>>>(QKV, GKV, OUTGLB);

  // 10) gate + fuse
  k_gate<<<(unsigned)((ML * Hc + 255) / 256), 256, 0, stream>>>(x, w_gate, b_gate, GATE);
  k_fuse<<<(unsigned)((ML * Dc + 255) / 256), 256, 0, stream>>>(OUTLOC, OUTGLB, GATE, FUSED);

  // 11) output projection straight into d_out
  cast(FUSED, FUSEDB, ML * Dc);
  gemm(FUSEDB, WT_OUT, b_out, out, (int)ML, Dc, Dc);
}
